// _Router_2027224563964
// MI455X (gfx1250) — matrix operand order repaired and checked by reading, not compiled, not run on hardware
//
#include <hip/hip_runtime.h>

#define TOKENS 16384
#define HIDDEN 4096
#define NEXP 64
#define TOPK 8
#define TOK_PER_BLOCK 128
#define THREADS 256

#define CHUNK_K 256                 // K elements staged per LDS buffer
#define KSTEPS  (CHUNK_K / 32)      // WMMA steps per chunk = 8
#define NCHUNK  (HIDDEN / CHUNK_K)  // 16
#define BPAD    (CHUNK_K + 8)       // padded halves per expert row (bank spread)

typedef __attribute__((ext_vector_type(16))) _Float16 v16h;
typedef __attribute__((ext_vector_type(8)))  _Float16 v8h;
typedef __attribute__((ext_vector_type(8)))  float    v8f;
typedef __attribute__((ext_vector_type(4)))  float    v4f;

// A fragment (16-bit A 16x32 layout): lane-half h starts at kc + h*8;
// halves [0..7] = K kb..kb+7, halves [8..15] = K kb+16..kb+23.
__device__ __forceinline__ v16h load_a_frag(const float* __restrict__ row, int kb) {
  v4f x0 = *(const v4f*)(row + kb);
  v4f x1 = *(const v4f*)(row + kb + 4);
  v4f x2 = *(const v4f*)(row + kb + 16);
  v4f x3 = *(const v4f*)(row + kb + 20);
  v16h f;
#pragma unroll
  for (int i = 0; i < 4; ++i) {
    f[i]      = (_Float16)x0[i];
    f[4 + i]  = (_Float16)x1[i];
    f[8 + i]  = (_Float16)x2[i];
    f[12 + i] = (_Float16)x3[i];
  }
  return f;
}

// B fragment from LDS (32x16 B layout): lane-half h holds K klocal+h*8..+7
// then klocal+16+h*8..+7 of its expert's row. Two aligned 16B LDS loads.
__device__ __forceinline__ v16h load_b_frag_lds(const _Float16* __restrict__ bp) {
  v8h lo = *(const v8h*)(bp);
  v8h hi = *(const v8h*)(bp + 16);
  v16h f;
#pragma unroll
  for (int j = 0; j < 8; ++j) { f[j] = lo[j]; f[8 + j] = hi[j]; }
  return f;
}

__global__ __launch_bounds__(THREADS)
void router_wmma_kernel(const float* __restrict__ hs,
                        const float* __restrict__ wt,
                        float* __restrict__ scores_out,
                        int* __restrict__ idx_out) {
  __shared__ _Float16 s_b[2][NEXP * BPAD];       // 2 x 33792 B f16 weight buffers
  __shared__ float    s_log[TOK_PER_BLOCK * 65]; // stride 65 -> conflict-free rows
  __shared__ float    s_inv[TOK_PER_BLOCK];

  const int tid  = threadIdx.x;
  const int wave = tid >> 5;
  const int lane = tid & 31;
  const int half = lane >> 4;
  const int l16  = lane & 15;

  const int tokBase = blockIdx.x * TOK_PER_BLOCK + wave * 16;
  const float* arow = hs + (size_t)(tokBase + l16) * HIDDEN;

  // --- weight staging mapping: 4 threads per expert, 64 k each ---
  const int se = tid >> 2;          // expert 0..63
  const int sk = (tid & 3) * 64;    // k offset inside chunk
  const float* srow = wt + (size_t)se * HIDDEN + sk;
  _Float16* sdst_base = &s_b[0][0]; (void)sdst_base;

  v4f r[16];  // staging registers: 64 f32 weight elements

  // prologue: stage chunk 0
#pragma unroll
  for (int i = 0; i < 16; ++i) r[i] = *(const v4f*)(srow + 4 * i);
  {
    _Float16* dst = &s_b[0][se * BPAD + sk];
#pragma unroll
    for (int i = 0; i < 8; ++i) {
      v8h t;
#pragma unroll
      for (int j = 0; j < 4; ++j) {
        t[j]     = (_Float16)r[2 * i][j];
        t[4 + j] = (_Float16)r[2 * i + 1][j];
      }
      *(v8h*)(dst + 8 * i) = t;
    }
  }
  __syncthreads();

  v8f acc0 = {}, acc1 = {}, acc2 = {}, acc3 = {};

  for (int c = 0; c < NCHUNK; ++c) {
    const int cur = c & 1;

    // issue next chunk's global loads early (latency overlapped with compute)
    if (c + 1 < NCHUNK) {
      const float* src = srow + (c + 1) * CHUNK_K;
#pragma unroll
      for (int i = 0; i < 16; ++i) r[i] = *(const v4f*)(src + 4 * i);
    }

    // 8 WMMA steps out of the current LDS weight buffer
    const _Float16* bbase = &s_b[cur][(size_t)l16 * BPAD + half * 8];
#pragma unroll
    for (int s = 0; s < KSTEPS; ++s) {
      const int kc = c * CHUNK_K + s * 32;
      v16h a  = load_a_frag(arow, kc + half * 8);
      const _Float16* bp = bbase + s * 32;
      v16h b0 = load_b_frag_lds(bp);
      v16h b1 = load_b_frag_lds(bp + 16 * BPAD);
      v16h b2 = load_b_frag_lds(bp + 32 * BPAD);
      v16h b3 = load_b_frag_lds(bp + 48 * BPAD);
      acc0 = __builtin_amdgcn_wmma_f32_16x16x32_f16(false, a, false, b0, (short)0, acc0, false, false);
      acc1 = __builtin_amdgcn_wmma_f32_16x16x32_f16(false, a, false, b1, (short)0, acc1, false, false);
      acc2 = __builtin_amdgcn_wmma_f32_16x16x32_f16(false, a, false, b2, (short)0, acc2, false, false);
      acc3 = __builtin_amdgcn_wmma_f32_16x16x32_f16(false, a, false, b3, (short)0, acc3, false, false);
    }

    // convert + store next chunk into the other buffer
    if (c + 1 < NCHUNK) {
      _Float16* dst = &s_b[cur ^ 1][se * BPAD + sk];
#pragma unroll
      for (int i = 0; i < 8; ++i) {
        v8h t;
#pragma unroll
        for (int j = 0; j < 4; ++j) {
          t[j]     = (_Float16)r[2 * i][j];
          t[4 + j] = (_Float16)r[2 * i + 1][j];
        }
        *(v8h*)(dst + 8 * i) = t;
      }
    }
    __syncthreads();
  }

  // Spill logits: C/D layout — VGPR r holds M=r (lanes 0-15) / M=r+8 (lanes 16-31), N=l16.
  {
    const int tlBase = wave * 16 + half * 8;
#pragma unroll
    for (int rr = 0; rr < 8; ++rr) {
      float* dst = &s_log[(tlBase + rr) * 65];
      dst[ 0 + l16] = acc0[rr];
      dst[16 + l16] = acc1[rr];
      dst[32 + l16] = acc2[rr];
      dst[48 + l16] = acc3[rr];
    }
  }
  __syncthreads();

  // Per-token softmax (one thread per token); keep unnormalized probs in LDS.
  if (tid < TOK_PER_BLOCK) {
    float* row = &s_log[tid * 65];
    float mx = row[0];
#pragma unroll
    for (int e = 1; e < NEXP; ++e) mx = fmaxf(mx, row[e]);
    float sum = 0.f;
#pragma unroll
    for (int e = 0; e < NEXP; ++e) {
      float p = __expf(row[e] - mx);
      row[e] = p;
      sum += p;
    }
    s_inv[tid] = 1.0f / sum;
  }
  __syncthreads();

  // Coalesced score stores by all 256 threads.
  {
    const size_t outBase = (size_t)blockIdx.x * TOK_PER_BLOCK * NEXP;
    for (int i = tid; i < TOK_PER_BLOCK * NEXP; i += THREADS) {
      const int tl = i >> 6;
      const int e  = i & 63;
      scores_out[outBase + (size_t)i] = s_log[tl * 65 + e] * s_inv[tl];
    }
  }
  __syncthreads();

  // Top-8 by 8-pass argmax; strict '>' keeps earliest index on ties (matches
  // jax.lax.top_k). Destroys LDS probs (safe: scores already stored).
  if (tid < TOK_PER_BLOCK) {
    float* row = &s_log[tid * 65];
    const size_t token = (size_t)blockIdx.x * TOK_PER_BLOCK + tid;
#pragma unroll
    for (int k = 0; k < TOPK; ++k) {
      float best = -3.402823466e38f;
      int bi = 0;
      for (int e = 0; e < NEXP; ++e) {
        float v = row[e];
        if (v > best) { best = v; bi = e; }
      }
      idx_out[token * TOPK + k] = bi;
      row[bi] = -3.402823466e38f;
    }
  }
}

extern "C" void kernel_launch(void* const* d_in, const int* in_sizes, int n_in,
                              void* d_out, int out_size, void* d_ws, size_t ws_size,
                              hipStream_t stream) {
  const float* hs = (const float*)d_in[0];   // [16384, 4096] f32
  const float* wt = (const float*)d_in[1];   // [64, 4096] f32
  float* scores  = (float*)d_out;                            // [16384, 64] f32
  int*   indices = (int*)(scores + (size_t)TOKENS * NEXP);   // [16384, 8] i32

  dim3 grid(TOKENS / TOK_PER_BLOCK);
  dim3 block(THREADS);
  hipLaunchKernelGGL(router_wmma_kernel, grid, block, 0, stream,
                     hs, wt, scores, indices);
}
